// CVCL_loss_84069689852374
// MI455X (gfx1250) — matrix operand order repaired and checked by reading, not compiled, not run on hardware
//
#include <hip/hip_runtime.h>
#include <hip/hip_bf16.h>

// Problem constants (from the reference)
#define BATCH  32
#define NSAMP  16384     // S
#define NCLUS  64        // K
#define NROWS  128       // N = 2K
#define KSPLIT 4         // S-dimension split for the Gram kernel
#define SCHUNK (NSAMP / KSPLIT)

typedef __attribute__((ext_vector_type(16))) __bf16        v16bf;
typedef __attribute__((ext_vector_type(8)))  float         v8f;
typedef __attribute__((ext_vector_type(4)))  unsigned int  u32x4;
typedef __attribute__((ext_vector_type(8)))  unsigned int  u32x8;

__device__ __forceinline__ unsigned short f2bf(float f) {
    unsigned int u = __float_as_uint(f);
    u += 0x7FFFu + ((u >> 16) & 1u);   // round-to-nearest-even
    return (unsigned short)(u >> 16);
}

__device__ __forceinline__ v16bf load_op16(const unsigned short* p) {
    u32x4 lo = *(const u32x4*)p;        // 8 bf16
    u32x4 hi = *(const u32x4*)(p + 8);  // next 8 bf16 (16B-aligned)
    u32x8 r = {lo[0], lo[1], lo[2], lo[3], hi[0], hi[1], hi[2], hi[3]};
    return __builtin_bit_cast(v16bf, r);
}

// ---------------------------------------------------------------------------
// Kernel 1: column sums cs[b][src][k] = sum_s q[b,s,k]
// grid = (2, B), block = 256; element index = tid + 256*i (fully coalesced).
// ---------------------------------------------------------------------------
__global__ void __launch_bounds__(256)
colsum_kernel(const float* __restrict__ q_i, const float* __restrict__ q_j,
              float* __restrict__ cs) {
    const int src = blockIdx.x;            // 0 = q_i, 1 = q_j
    const int b   = blockIdx.y;
    const int tid = threadIdx.x;
    const float* __restrict__ q =
        (src ? q_j : q_i) + (size_t)b * NSAMP * NCLUS;

    float acc = 0.0f;
    for (int i = 0; i < NSAMP / 4; ++i)
        acc += q[(size_t)i * 256 + tid];

    __shared__ float red[256];
    red[tid] = acc;
    __syncthreads();
    if (tid < NCLUS) {
        float s = red[tid] + red[tid + 64] + red[tid + 128] + red[tid + 192];
        cs[((size_t)b * 2 + src) * NCLUS + tid] = s;
    }
}

// ---------------------------------------------------------------------------
// Kernel 2: P[b][src*64 + k][s] = bf16( (q[s,k]^2 / cs[k]) / rowsum )
// grid = (S/256, 2, B), block = 256; one sample row per thread.
// P is row-major [B][128][S] bf16; per-k stores are coalesced across threads.
// ---------------------------------------------------------------------------
__global__ void __launch_bounds__(256)
pnorm_kernel(const float* __restrict__ q_i, const float* __restrict__ q_j,
             const float* __restrict__ cs, unsigned short* __restrict__ P) {
    const int s   = blockIdx.x * 256 + threadIdx.x;
    const int src = blockIdx.y;
    const int b   = blockIdx.z;

    const float* __restrict__ qrow =
        (src ? q_j : q_i) + ((size_t)b * NSAMP + s) * NCLUS;
    const float* __restrict__ csb = cs + ((size_t)b * 2 + src) * NCLUS;

    float w[NCLUS];
    float sum = 0.0f;
#pragma unroll
    for (int k = 0; k < NCLUS; k += 4) {
        float4 v = *(const float4*)(qrow + k);
        float w0 = v.x * v.x / csb[k + 0];
        float w1 = v.y * v.y / csb[k + 1];
        float w2 = v.z * v.z / csb[k + 2];
        float w3 = v.w * v.w / csb[k + 3];
        w[k + 0] = w0; w[k + 1] = w1; w[k + 2] = w2; w[k + 3] = w3;
        sum += (w0 + w1) + (w2 + w3);
    }
    const float inv = 1.0f / sum;

    unsigned short* __restrict__ prow =
        P + ((size_t)b * NROWS + (size_t)src * NCLUS) * NSAMP + s;
#pragma unroll
    for (int k = 0; k < NCLUS; ++k)
        prow[(size_t)k * NSAMP] = f2bf(w[k] * inv);
}

// ---------------------------------------------------------------------------
// Kernel 3: simP[split][b] += P[b] * P[b]^T over an S-chunk, 2x2 register
// tiling: each wave owns a 32x32 super-tile (4 f32 accumulators) and per
// K-step loads A0/A1/B0/B1 once -> 8 b128 loads feed 4 WMMAs (2 loads/WMMA,
// half the L2 traffic of the 1-tile version). grid = (16, KSPLIT, B),
// block = 32 (one wave, EXEC all 1s as WMMA requires).
// Operand layouts used here (contiguous k, not the ISA 7.12.2 table order):
//   A (16x32 bf16): lanes 0-15 row m K{0..15};       lanes 16-31 K{16..31}
//   B (32x16 bf16): lanes 0-15 col n K{0..15};       lanes 16-31 K{16..31}
// B-tile == P rows of the n-block, so both operands read row-major P.
// ---------------------------------------------------------------------------
__global__ void __launch_bounds__(32)
gram_kernel(const unsigned short* __restrict__ P, float* __restrict__ simP) {
    const int lane  = threadIdx.x;
    const int grp   = blockIdx.x;          // 0..15 -> 4x4 grid of 32x32 tiles
    const int split = blockIdx.y;          // 0..KSPLIT-1
    const int b     = blockIdx.z;
    const int gm    = (grp >> 2) * 32;
    const int gn    = (grp & 3) * 32;

    const unsigned short* __restrict__ Pb = P + (size_t)b * NROWS * NSAMP;
    const unsigned short* __restrict__ rowA0 = Pb + (size_t)(gm +      (lane & 15)) * NSAMP;
    const unsigned short* __restrict__ rowA1 = Pb + (size_t)(gm + 16 + (lane & 15)) * NSAMP;
    const unsigned short* __restrict__ rowB0 = Pb + (size_t)(gn +      (lane & 15)) * NSAMP;
    const unsigned short* __restrict__ rowB1 = Pb + (size_t)(gn + 16 + (lane & 15)) * NSAMP;
    const int aSel = (lane < 16) ? 0 : 16;   // half-elements
    const int bSel = (lane < 16) ? 0 : 16;   // half-elements

    v8f acc00 = {}, acc01 = {}, acc10 = {}, acc11 = {};
    const int sEnd = split * SCHUNK + SCHUNK;
    for (int s0 = split * SCHUNK; s0 < sEnd; s0 += 32) {
        // A chunks: two 16B loads each at k-offsets {sel, sel+8}
        u32x4 a0l = *(const u32x4*)(rowA0 + s0 + aSel);
        u32x4 a0h = *(const u32x4*)(rowA0 + s0 + aSel + 8);
        u32x4 a1l = *(const u32x4*)(rowA1 + s0 + aSel);
        u32x4 a1h = *(const u32x4*)(rowA1 + s0 + aSel + 8);
        // B chunks: 16 contiguous bf16 at k-offset sel
        v16bf bv0 = load_op16(rowB0 + s0 + bSel);
        v16bf bv1 = load_op16(rowB1 + s0 + bSel);
        u32x8 ar0 = {a0l[0], a0l[1], a0l[2], a0l[3], a0h[0], a0h[1], a0h[2], a0h[3]};
        u32x8 ar1 = {a1l[0], a1l[1], a1l[2], a1l[3], a1h[0], a1h[1], a1h[2], a1h[3]};
        v16bf av0 = __builtin_bit_cast(v16bf, ar0);
        v16bf av1 = __builtin_bit_cast(v16bf, ar1);

        acc00 = __builtin_amdgcn_wmma_f32_16x16x32_bf16(false, av0, false, bv0,
                                                        (short)0, acc00, false, false);
        acc01 = __builtin_amdgcn_wmma_f32_16x16x32_bf16(false, av0, false, bv1,
                                                        (short)0, acc01, false, false);
        acc10 = __builtin_amdgcn_wmma_f32_16x16x32_bf16(false, av1, false, bv0,
                                                        (short)0, acc10, false, false);
        acc11 = __builtin_amdgcn_wmma_f32_16x16x32_bf16(false, av1, false, bv1,
                                                        (short)0, acc11, false, false);
    }

    // C/D layout: VGPR r, lanes 0-15 -> M=r, N=lane; lanes 16-31 -> M=8+r.
    float* __restrict__ out =
        simP + (((size_t)split * BATCH + b) * NROWS + gm) * NROWS + gn;
    const int n    = lane & 15;
    const int mOff = (lane < 16) ? 0 : 8;
#pragma unroll
    for (int r = 0; r < 8; ++r) {
        out[(mOff + r) * NROWS + n]            = acc00[r];
        out[(mOff + r) * NROWS + n + 16]       = acc01[r];
        out[(mOff + r + 16) * NROWS + n]       = acc10[r];
        out[(mOff + r + 16) * NROWS + n + 16]  = acc11[r];
    }
}

// ---------------------------------------------------------------------------
// Kernel 4: per-batch loss. Row r logits = (sum of K-split partials)/T over
// all columns except the diagonal; positive logit at column partner(r).
// loss_b = mean_r(LSE - pos). grid = B, block = 128.
// ---------------------------------------------------------------------------
__global__ void __launch_bounds__(128)
rowloss_kernel(const float* __restrict__ simP, const float* __restrict__ tptr,
               float* __restrict__ bloss) {
    const int b = blockIdx.x;
    const int r = threadIdx.x;
    const float invT = 1.0f / *tptr;
    const size_t rowOff = ((size_t)b * NROWS + r) * NROWS;
    const float* __restrict__ p0 = simP + rowOff;
    const size_t step = (size_t)BATCH * NROWS * NROWS;

    float rowv[NROWS];
#pragma unroll 4
    for (int c = 0; c < NROWS; ++c) {
        float v = p0[c] + p0[c + step] + p0[c + 2 * step] + p0[c + 3 * step];
        rowv[c] = v * invT;
    }

    float m = -3.4e38f;
    for (int c = 0; c < NROWS; ++c)
        if (c != r) m = fmaxf(m, rowv[c]);
    float se = 0.0f;
    for (int c = 0; c < NROWS; ++c)
        if (c != r) se += __expf(rowv[c] - m);
    const float lse = m + __logf(se);
    const int partner = (r < NCLUS) ? r + NCLUS : r - NCLUS;
    float v = lse - rowv[partner];

    __shared__ float red[NROWS];
    red[r] = v;
    __syncthreads();
    for (int off = NROWS / 2; off > 0; off >>= 1) {
        if (r < off) red[r] += red[r + off];
        __syncthreads();
    }
    if (r == 0) bloss[b] = red[0] * (1.0f / (float)NROWS);
}

// ---------------------------------------------------------------------------
// Kernel 5: mean over batches -> d_out[0] (always written; d_out is poisoned).
// ---------------------------------------------------------------------------
__global__ void finalize_kernel(const float* __restrict__ bloss,
                                float* __restrict__ out) {
    if (threadIdx.x == 0) {
        float s = 0.0f;
        for (int i = 0; i < BATCH; ++i) s += bloss[i];
        out[0] = s * (1.0f / (float)BATCH);
    }
}

extern "C" void kernel_launch(void* const* d_in, const int* in_sizes, int n_in,
                              void* d_out, int out_size, void* d_ws, size_t ws_size,
                              hipStream_t stream) {
    const float* q_i  = (const float*)d_in[0];
    const float* q_j  = (const float*)d_in[1];
    const float* tptr = (const float*)d_in[2];
    float*       out  = (float*)d_out;

    // Workspace layout (all 256B aligned):
    //   [0, 16KB)            : column sums   [B][2][64] f32
    //   [16KB, 16KB+128MB)   : P bf16        [B][128][16384]
    //   next 8MB             : sim partials  [KSPLIT][B][128][128] f32
    //   next 128B            : per-batch losses [B] f32
    char* ws = (char*)d_ws;
    float*          cs   = (float*)ws;
    unsigned short* P    = (unsigned short*)(ws + (16 << 10));
    const size_t    pOff = (size_t)(16 << 10) +
                           (size_t)BATCH * NROWS * NSAMP * sizeof(unsigned short);
    float*          simP = (float*)(ws + pOff);
    float*          bls  = (float*)(ws + pOff +
                           (size_t)KSPLIT * BATCH * NROWS * NROWS * sizeof(float));

    colsum_kernel<<<dim3(2, BATCH), 256, 0, stream>>>(q_i, q_j, cs);
    pnorm_kernel<<<dim3(NSAMP / 256, 2, BATCH), 256, 0, stream>>>(q_i, q_j, cs, P);
    gram_kernel<<<dim3(16, KSPLIT, BATCH), 32, 0, stream>>>(P, simP);
    rowloss_kernel<<<dim3(BATCH), NROWS, 0, stream>>>(simP, tptr, bls);
    finalize_kernel<<<1, 32, 0, stream>>>(bls, out);
}
